// QuantizedLinner_4750233829798
// MI455X (gfx1250) — 8-bit matrix operand order repaired and checked by reading, compiled, not run on hardware
//
#include <hip/hip_runtime.h>
#include <hip/hip_bf16.h>

typedef __attribute__((ext_vector_type(8))) int v8i;

constexpr int B_DIM = 8192;
constexpr int IN_F  = 4096;
constexpr int OUT_F = 4096;
constexpr float INPUT_SCALE  = 0.05f;
constexpr float OUTPUT_SCALE = 0.1f;

constexpr int LDS_ROW   = 80;
constexpr int A_ROWS    = 128;
constexpr int B_ROWS    = 256;
constexpr int A_BYTES   = A_ROWS * LDS_ROW;
constexpr int BUF_BYTES = (A_ROWS + B_ROWS) * LDS_ROW;

__device__ __forceinline__ void async_copy16(const void* gsrc, unsigned lds_off) {
    asm volatile("global_load_async_to_lds_b128 %0, %1, off"
                 :: "v"(lds_off), "v"(gsrc) : "memory");
}
__device__ __forceinline__ void wait_async0() {
    asm volatile("s_wait_asynccnt 0x0" ::: "memory");
}

__global__ void __launch_bounds__(256)
pack_i32_to_i8(const int* __restrict__ src, unsigned int* __restrict__ dst, int n4) {
    int idx = blockIdx.x * blockDim.x + threadIdx.x;
    if (idx >= n4) return;
    int4 v = ((const int4*)src)[idx];
    unsigned int packed = (v.x & 0xFF)
                        | ((v.y & 0xFF) << 8)
                        | ((v.z & 0xFF) << 16)
                        | ((v.w & 0xFF) << 24);
    dst[idx] = packed;
}

__global__ void __launch_bounds__(256, 1)
qlinear_wmma_iu8(const signed char* __restrict__ Xp,
                 const signed char* __restrict__ Wp,
                 const int* __restrict__ bias,
                 const float* __restrict__ wscale,
                 float* __restrict__ out)
{
    __shared__ alignas(16) unsigned char smem[2][BUF_BYTES];

    const int t     = threadIdx.x;
    const int lane  = t & 31;
    const int wave  = t >> 5;
    const int waveM = wave >> 2;
    const int waveN = wave & 3;

    const int mBlock = blockIdx.y * 128;
    const int nBlock = blockIdx.x * 256;
    const int mBase  = mBlock + waveM * 64;
    const int nBase  = nBlock + waveN * 64;

    const int g = lane >> 4;
    const int l = lane & 15;

    const unsigned lds0 = (unsigned)(size_t)&smem[0][0];
    const unsigned lds1 = (unsigned)(size_t)&smem[1][0];

    auto issue_stage = [&](unsigned ldsBuf, int k0) {
#pragma unroll
        for (int c = 0; c < 2; ++c) {
            int idx = c * 256 + t;
            int row = idx >> 2, off = (idx & 3) * 16;
            async_copy16(Xp + (size_t)(mBlock + row) * IN_F + k0 + off,
                         ldsBuf + (unsigned)(row * LDS_ROW + off));
        }
#pragma unroll
        for (int c = 0; c < 4; ++c) {
            int idx = c * 256 + t;
            int row = idx >> 2, off = (idx & 3) * 16;
            async_copy16(Wp + (size_t)(nBlock + row) * IN_F + k0 + off,
                         ldsBuf + (unsigned)(A_BYTES + row * LDS_ROW + off));
        }
    };

    v8i acc[4][4];
#pragma unroll
    for (int i = 0; i < 4; ++i)
#pragma unroll
        for (int j = 0; j < 4; ++j)
            acc[i][j] = (v8i)0;

    constexpr int NK = IN_F / 64;

    issue_stage(lds0, 0);
    wait_async0();
    __syncthreads();

    int cur = 0;
    for (int ks = 0; ks < NK; ++ks) {
        if (ks + 1 < NK)
            issue_stage(cur ? lds0 : lds1, (ks + 1) * 64);

        const unsigned char* Abuf = &smem[cur][0];
        const unsigned char* Bbuf = &smem[cur][A_BYTES];

        v8i a[4];
#pragma unroll
        for (int i = 0; i < 4; ++i) {
            const unsigned char* ar = Abuf + (waveM * 64 + i * 16 + l) * LDS_ROW + g * 8;
            int2 q0 = *(const int2*)(ar + 0);
            int2 q1 = *(const int2*)(ar + 16);
            int2 q2 = *(const int2*)(ar + 32);
            int2 q3 = *(const int2*)(ar + 48);
            a[i] = (v8i){q0.x, q0.y, q1.x, q1.y, q2.x, q2.y, q3.x, q3.y};
        }

        v8i b[4];
#pragma unroll
        for (int j = 0; j < 4; ++j) {
            const unsigned char* br = Bbuf + (waveN * 64 + j * 16 + l) * LDS_ROW + g * 8;
            int2 r0 = *(const int2*)(br + 0);
            int2 r1 = *(const int2*)(br + 16);
            int2 r2 = *(const int2*)(br + 32);
            int2 r3 = *(const int2*)(br + 48);
            b[j] = (v8i){r0.x, r0.y, r1.x, r1.y, r2.x, r2.y, r3.x, r3.y};
        }

#pragma unroll
        for (int i = 0; i < 4; ++i)
#pragma unroll
            for (int j = 0; j < 4; ++j)
                acc[i][j] = __builtin_amdgcn_wmma_i32_16x16x64_iu8(
                    true, a[i], true, b[j], acc[i][j], false, false);

        if (ks + 1 < NK) {
            wait_async0();
            __syncthreads();
            cur ^= 1;
        }
    }

    const float base_scale = INPUT_SCALE / OUTPUT_SCALE;
#pragma unroll
    for (int j = 0; j < 4; ++j) {
        const int n   = nBase + j * 16 + l;
        const float s = base_scale * wscale[n];
        const int  bi = bias[n];
#pragma unroll
        for (int i = 0; i < 4; ++i) {
            const int mSub = mBase + i * 16 + g * 8;
#pragma unroll
            for (int r = 0; r < 8; ++r) {
                float v = (float)(acc[i][j][r] + bi) * s;
                v = rintf(v);
                v = fminf(fmaxf(v, -128.0f), 127.0f);
                out[(size_t)(mSub + r) * OUT_F + n] = v;
            }
        }
    }
}

extern "C" void kernel_launch(void* const* d_in, const int* in_sizes, int n_in,
                              void* d_out, int out_size, void* d_ws, size_t ws_size,
                              hipStream_t stream) {
    const int*   x_i32  = (const int*)d_in[0];
    const int*   w_i32  = (const int*)d_in[1];
    const int*   bias   = (const int*)d_in[2];
    const float* wscale = (const float*)d_in[3];
    float*       out    = (float*)d_out;

    signed char* xp = (signed char*)d_ws;
    signed char* wp = xp + (size_t)B_DIM * IN_F;

    {
        int n4 = (B_DIM * IN_F) / 4;
        pack_i32_to_i8<<<(n4 + 255) / 256, 256, 0, stream>>>(x_i32, (unsigned int*)xp, n4);
    }
    {
        int n4 = (OUT_F * IN_F) / 4;
        pack_i32_to_i8<<<(n4 + 255) / 256, 256, 0, stream>>>(w_i32, (unsigned int*)wp, n4);
    }

    dim3 grid(OUT_F / 256, B_DIM / 128);
    qlinear_wmma_iu8<<<grid, 256, 0, stream>>>(xp, wp, bias, wscale, out);
}
